// PointNetPPReg_67585605370628
// MI455X (gfx1250) — matrix operand order repaired and checked by reading, not compiled, not run on hardware
//
#include <hip/hip_runtime.h>
#include <hip/hip_bf16.h>
#include <stdint.h>
#include <stddef.h>

// ---------------------------------------------------------------------------
// PointNet++ regression forward for MI455X (gfx1250, wave32, WMMA).
// - All MLP GEMMs: bf16 WMMA (v_wmma_f32_16x16x32_bf16), f32 accumulate,
//   bf16 activation storage (memory-bound workload -> halve HBM traffic).
// - Bias + ReLU + masked BatchNorm statistics fused into the GEMM epilogue.
// - BatchNorm APPLICATION folded into the NEXT layer's weights/bias:
//     (Y*sc + sh) @ W = Y @ (diag(sc) W) + (sh.W + b)
//   so no elementwise normalize pass ever touches the big activation tensors.
// - GEMM inner loop is software-pipelined: next k-step's A/B fragments are
//   loaded before the current k-step's WMMAs, so loads overlap the matrix pipe
//   (steady state waits: s_wait_loadcnt 8/6/4/2, not 0).
// - FPS: one workgroup per cloud, dist[] in LDS, shuffle argmax reduction.
// - Ball query: one wave per center, ballot+popc ordered compaction (first K
//   in index order == reference's sort-based selection).
// ---------------------------------------------------------------------------

typedef __attribute__((ext_vector_type(16))) __bf16 v16bf;
typedef __attribute__((ext_vector_type(8)))  __bf16 v8bf;
typedef __attribute__((ext_vector_type(8)))  float  v8f;
typedef __attribute__((ext_vector_type(4)))  int    v4i;

#define B_   8
#define N_   4096
#define K_   64
#define M1_  2048
#define M2_  512
#define E1_  (B_ * M1_ * K_)   // 1048576
#define E2_  (B_ * M2_ * K_)   // 262144
#define E3_  (B_ * M2_)        // 4096
#define EPS_ 1e-5f

// ---------------------------------------------------------------------------
// Utility kernels
// ---------------------------------------------------------------------------
__global__ void zero_u32_kernel(unsigned* __restrict__ p, int n) {
  int i = blockIdx.x * blockDim.x + threadIdx.x;
  if (i < n) p[i] = 0u;
}

// ---------------------------------------------------------------------------
// Farthest point sampling: one workgroup per batch, dist[] in LDS.
// ---------------------------------------------------------------------------
__global__ __launch_bounds__(256) void fps_kernel(const float* __restrict__ pos,
                                                  int n, int m,
                                                  int* __restrict__ idx_out) {
  const int b = blockIdx.x;
  const float* p = pos + (size_t)b * n * 3;
  int* out = idx_out + (size_t)b * m;
  __shared__ float dist[N_];
  __shared__ float wv[8];
  __shared__ int   wi[8];
  __shared__ int   s_sel;
  const int tid  = threadIdx.x;
  const int nt   = blockDim.x;
  const int lane = tid & 31;
  const int wid  = tid >> 5;

  const float x0 = p[0], y0 = p[1], z0 = p[2];
  for (int i = tid; i < n; i += nt) {
    float dx = p[3 * i + 0] - x0, dy = p[3 * i + 1] - y0, dz = p[3 * i + 2] - z0;
    dist[i] = dx * dx + dy * dy + dz * dz;
  }
  if (tid == 0) out[0] = 0;
  __syncthreads();

  for (int it = 1; it < m; ++it) {
    // argmax(dist) with first-occurrence (lowest index) tie-break
    float bv = -1.0f; int bi = 0;
    for (int i = tid; i < n; i += nt) {
      float d = dist[i];
      if (d > bv) { bv = d; bi = i; }
    }
    for (int off = 16; off > 0; off >>= 1) {
      float ov = __shfl_down(bv, off, 32);
      int   oi = __shfl_down(bi, off, 32);
      if (ov > bv || (ov == bv && oi < bi)) { bv = ov; bi = oi; }
    }
    if (lane == 0) { wv[wid] = bv; wi[wid] = bi; }
    __syncthreads();
    if (tid == 0) {
      float best = wv[0]; int besti = wi[0];
      for (int w = 1; w < nt / 32; ++w)
        if (wv[w] > best || (wv[w] == best && wi[w] < besti)) { best = wv[w]; besti = wi[w]; }
      s_sel = besti;
      out[it] = besti;
    }
    __syncthreads();
    const int sel = s_sel;
    const float sx = p[3 * sel], sy = p[3 * sel + 1], sz = p[3 * sel + 2];
    for (int i = tid; i < n; i += nt) {
      float dx = p[3 * i] - sx, dy = p[3 * i + 1] - sy, dz = p[3 * i + 2] - sz;
      float nd = dx * dx + dy * dy + dz * dz;
      if (nd < dist[i]) dist[i] = nd;
    }
    __syncthreads();
  }
}

__global__ void gather_centers_kernel(const float* __restrict__ pos,
                                      const int* __restrict__ idx,
                                      float* __restrict__ centers,
                                      int n, int mTot, int M) {
  int i = blockIdx.x * blockDim.x + threadIdx.x;
  if (i >= mTot) return;
  int b = i / M;
  int j = idx[i];
  const float* p = pos + ((size_t)b * n + j) * 3;
  centers[(size_t)i * 3 + 0] = p[0];
  centers[(size_t)i * 3 + 1] = p[1];
  centers[(size_t)i * 3 + 2] = p[2];
}

// ---------------------------------------------------------------------------
// Ball query: one wave per center; first K in-radius points in index order.
// ---------------------------------------------------------------------------
__global__ __launch_bounds__(256) void radius_kernel(const float* __restrict__ pos,
                                                     const float* __restrict__ centers,
                                                     float r2, int n, int mTot, int M,
                                                     int* __restrict__ nbr,
                                                     int* __restrict__ mask,
                                                     int* __restrict__ cntTotal) {
  const int c = blockIdx.x * (blockDim.x >> 5) + (threadIdx.x >> 5);
  const int lane = threadIdx.x & 31;
  if (c >= mTot) return;
  const int b = c / M;
  const float* p = pos + (size_t)b * n * 3;
  const float cx = centers[(size_t)c * 3 + 0];
  const float cy = centers[(size_t)c * 3 + 1];
  const float cz = centers[(size_t)c * 3 + 2];
  int cnt = 0;
  for (int basej = 0; basej < n && cnt < K_; basej += 32) {
    const int j = basej + lane;   // n is a multiple of 32
    float dx = p[3 * j + 0] - cx, dy = p[3 * j + 1] - cy, dz = p[3 * j + 2] - cz;
    bool pred = (dx * dx + dy * dy + dz * dz) <= r2;
    unsigned mb = (unsigned)__ballot(pred);
    int posIn = cnt + __popc(mb & ((1u << lane) - 1u));
    if (pred && posIn < K_) nbr[(size_t)c * K_ + posIn] = j;
    cnt += __popc(mb);
  }
  if (cnt > K_) cnt = K_;
  for (int k = lane; k < K_; k += 32) {
    int v = (k < cnt) ? 1 : 0;
    mask[(size_t)c * K_ + k] = v;
    if (!v) nbr[(size_t)c * K_ + k] = 0;
  }
  if (lane == 0) atomicAdd(cntTotal, cnt);
}

// ---------------------------------------------------------------------------
// Edge-feature builders (write bf16, K-padded to multiple of 32)
// ---------------------------------------------------------------------------
__global__ void build_x1_kernel(const float* __restrict__ x, const float* __restrict__ pos,
                                const float* __restrict__ centers, const int* __restrict__ nbr,
                                __bf16* __restrict__ X) {
  int e = blockIdx.x * blockDim.x + threadIdx.x;
  if (e >= E1_) return;
  int m = e / K_;           // global center index
  int b = m / M1_;
  int j = nbr[e];
  const float* p  = pos + ((size_t)b * N_ + j) * 3;
  const float* cc = centers + (size_t)m * 3;
  __bf16* xo = X + (size_t)e * 32;
  xo[0] = (__bf16)x[(size_t)b * N_ + j];
  xo[1] = (__bf16)(p[0] - cc[0]);
  xo[2] = (__bf16)(p[1] - cc[1]);
  xo[3] = (__bf16)(p[2] - cc[2]);
#pragma unroll
  for (int t = 4; t < 32; ++t) xo[t] = (__bf16)0.f;
}

__global__ void build_x2_kernel(const float* __restrict__ feat, const float* __restrict__ pos1,
                                const float* __restrict__ centers2, const int* __restrict__ nbr,
                                __bf16* __restrict__ X) {
  int idx = blockIdx.x * blockDim.x + threadIdx.x;
  if (idx >= E2_ * 5) return;
  int e = idx / 5, cg = idx - e * 5;
  int m = e / K_;
  int b = m / M2_;
  int j = nbr[e];
  const float* pj = pos1 + ((size_t)b * M1_ + j) * 3;
  const float* cc = centers2 + (size_t)m * 3;
  __bf16* xo = X + (size_t)e * 160 + cg * 32;
  for (int t = 0; t < 32; ++t) {
    int ch = cg * 32 + t;
    float v;
    if (ch < 128)      v = feat[((size_t)b * M1_ + j) * 128 + ch];
    else if (ch < 131) v = pj[ch - 128] - cc[ch - 128];
    else               v = 0.f;
    xo[t] = (__bf16)v;
  }
}

__global__ void build_x3_kernel(const float* __restrict__ feat, const float* __restrict__ centers2,
                                __bf16* __restrict__ X) {
  int idx = blockIdx.x * blockDim.x + threadIdx.x;
  if (idx >= E3_ * 9) return;
  int row = idx / 9, cg = idx - row * 9;
  __bf16* xo = X + (size_t)row * 288 + cg * 32;
  for (int t = 0; t < 32; ++t) {
    int ch = cg * 32 + t;
    float v;
    if (ch < 256)      v = feat[(size_t)row * 256 + ch];
    else if (ch < 259) v = centers2[(size_t)row * 3 + (ch - 256)];
    else               v = 0.f;
    xo[t] = (__bf16)v;
  }
}

// ---------------------------------------------------------------------------
// Weight pack: f32 [Cin, Cout] row-major -> bf16 WMMA B fragments, with the
// previous layer's BatchNorm scale folded in per input channel k:
//   Wp[k][n] = W[k][n] * scalePrev[k]     (scalePrev == nullptr -> 1.0)
// Packed index: ((kt*NT + nt)*32 + lane)*16 + e, element value
// W[kt*32 + (e>>3)*16 + (lane>>4)*8 + (e&7)][nt*16 + (lane&15)] (0 past Cin).
// ---------------------------------------------------------------------------
__global__ void pack_w_kernel(const float* __restrict__ W, __bf16* __restrict__ Wp,
                              const float* __restrict__ scalePrev,
                              int Cin, int Cout, int Kpad) {
  const int NT = Cout >> 4;
  const int total = (Kpad >> 5) * NT * 512;
  int i = blockIdx.x * blockDim.x + threadIdx.x;
  if (i >= total) return;
  int e  = i & 15;
  int L  = (i >> 4) & 31;
  int tile = i >> 9;             // kt*NT + nt
  int nt = tile % NT;
  int kt = tile / NT;
  int k = kt * 32 + (e >> 3) * 16 + (L >> 4) * 8 + (e & 7);
  int n = nt * 16 + (L & 15);
  float v = 0.f;
  if (k < Cin) {
    v = W[(size_t)k * Cout + n];
    if (scalePrev) v *= scalePrev[k];
  }
  Wp[i] = (__bf16)v;
}

// Fold the previous layer's BatchNorm shift into this layer's bias:
//   bias'[c] = b[c] + sum_k shiftPrev[k] * W[k][c]
__global__ void fold_bias_kernel(const float* __restrict__ W, const float* __restrict__ b,
                                 const float* __restrict__ shiftPrev, float* __restrict__ biasOut,
                                 int Cin, int Cout) {
  int c = blockIdx.x * blockDim.x + threadIdx.x;
  if (c >= Cout) return;
  float s = b[c];
  for (int k = 0; k < Cin; ++k) s += shiftPrev[k] * W[(size_t)k * Cout + c];
  biasOut[c] = s;
}

// ---------------------------------------------------------------------------
// WMMA GEMM: Y = relu(X @ W + b), bf16 in / bf16 out, f32 accumulate.
// Each wave: 16 rows x 64 cols (1 A fragment reused over 4 B fragments).
// Software-pipelined: iteration kt+1's 5 fragment loads are issued before the
// 4 WMMAs of iteration kt, so memory latency overlaps the matrix pipe.
// Epilogue accumulates masked per-channel sum / sumsq for BatchNorm stats.
// E multiple of 128, Cout multiple of 64, Kpad multiple of 32.
// ---------------------------------------------------------------------------
__global__ __launch_bounds__(256) void gemm_bf16_kernel(
    const __bf16* __restrict__ X, const __bf16* __restrict__ Wp,
    const float* __restrict__ bias, __bf16* __restrict__ Y,
    const int* __restrict__ mask, float* __restrict__ sums,
    int E, int Cout, int Kpad) {
  const int wave = threadIdx.x >> 5;
  const int lane = threadIdx.x & 31;
  const int lr   = lane & 15;
  const int h    = lane >> 4;
  const int row0 = (blockIdx.x * 8 + wave) * 16;
  const int col0 = blockIdx.y * 64;
  const int NT = Cout >> 4;
  const int KT = Kpad >> 5;

  __shared__ float s_sum[64];
  __shared__ float s_sq[64];
  if (threadIdx.x < 64) { s_sum[threadIdx.x] = 0.f; s_sq[threadIdx.x] = 0.f; }
  __syncthreads();

  v8f acc[4];
#pragma unroll
  for (int nt = 0; nt < 4; ++nt)
#pragma unroll
    for (int i = 0; i < 8; ++i) acc[nt][i] = 0.f;

  // A fragment (bf16 16x32): lane holds row (lane&15); its 16 K-values are two
  // contiguous 8-half runs at K = h*8 and K = 16 + h*8  ->  two 16B loads.
  const __bf16* xrow = X + (size_t)(row0 + lr) * Kpad;
  auto loadA = [&](int kt) -> v16bf {
    const v8bf* pa = (const v8bf*)(xrow + kt * 32 + h * 8);
    v8bf lo = pa[0];
    v8bf hi = pa[2];
    v16bf a;
#pragma unroll
    for (int i = 0; i < 8; ++i) { a[i] = lo[i]; a[i + 8] = hi[i]; }
    return a;
  };
  auto loadB = [&](int kt, int nt) -> v16bf {
    return *(const v16bf*)(Wp + ((size_t)(kt * NT + (col0 >> 4) + nt) * 32 + lane) * 16);
  };

  // Software pipeline: prologue loads kt=0, steady state preloads kt+1.
  v16bf a  = loadA(0);
  v16bf b0 = loadB(0, 0), b1 = loadB(0, 1), b2 = loadB(0, 2), b3 = loadB(0, 3);
  for (int kt = 0; kt < KT - 1; ++kt) {
    v16bf an  = loadA(kt + 1);
    v16bf bn0 = loadB(kt + 1, 0);
    v16bf bn1 = loadB(kt + 1, 1);
    v16bf bn2 = loadB(kt + 1, 2);
    v16bf bn3 = loadB(kt + 1, 3);
    acc[0] = __builtin_amdgcn_wmma_f32_16x16x32_bf16(false, a, false, b0, (short)0, acc[0], false, false);
    acc[1] = __builtin_amdgcn_wmma_f32_16x16x32_bf16(false, a, false, b1, (short)0, acc[1], false, false);
    acc[2] = __builtin_amdgcn_wmma_f32_16x16x32_bf16(false, a, false, b2, (short)0, acc[2], false, false);
    acc[3] = __builtin_amdgcn_wmma_f32_16x16x32_bf16(false, a, false, b3, (short)0, acc[3], false, false);
    a = an; b0 = bn0; b1 = bn1; b2 = bn2; b3 = bn3;
  }
  acc[0] = __builtin_amdgcn_wmma_f32_16x16x32_bf16(false, a, false, b0, (short)0, acc[0], false, false);
  acc[1] = __builtin_amdgcn_wmma_f32_16x16x32_bf16(false, a, false, b1, (short)0, acc[1], false, false);
  acc[2] = __builtin_amdgcn_wmma_f32_16x16x32_bf16(false, a, false, b2, (short)0, acc[2], false, false);
  acc[3] = __builtin_amdgcn_wmma_f32_16x16x32_bf16(false, a, false, b3, (short)0, acc[3], false, false);

  // Per-row validity (shared across the 4 col tiles). Wave-uniform branch on
  // mask presence; 8 consecutive mask words fetched as two b128 loads.
  float mkv[8];
  if (mask) {
    const v4i* mp = (const v4i*)(mask + row0 + h * 8);
    v4i m0 = mp[0];
    v4i m1 = mp[1];
#pragma unroll
    for (int r = 0; r < 4; ++r) mkv[r] = m0[r] ? 1.f : 0.f;
#pragma unroll
    for (int r = 0; r < 4; ++r) mkv[r + 4] = m1[r] ? 1.f : 0.f;
  } else {
#pragma unroll
    for (int r = 0; r < 8; ++r) mkv[r] = 1.f;
  }

  // Epilogue: bias, ReLU, store bf16, masked per-channel stats.
#pragma unroll
  for (int nt = 0; nt < 4; ++nt) {
    const int col = col0 + nt * 16 + lr;
    const float bc = bias[col];
    float cs = 0.f, cq = 0.f;
#pragma unroll
    for (int r = 0; r < 8; ++r) {
      const int row = row0 + r + h * 8;   // D layout: M = r + 8*(lane>>4)
      float y = acc[nt][r] + bc;
      y = y > 0.f ? y : 0.f;
      Y[(size_t)row * Cout + col] = (__bf16)y;
      cs += y * mkv[r];
      cq += y * y * mkv[r];
    }
    cs += __shfl_xor(cs, 16, 32);  // combine the two row-halves of the column
    cq += __shfl_xor(cq, 16, 32);
    if (h == 0) {
      atomicAdd(&s_sum[nt * 16 + lr], cs);
      atomicAdd(&s_sq[nt * 16 + lr], cq);
    }
  }
  __syncthreads();
  if (threadIdx.x < 64) {
    atomicAdd(&sums[col0 + threadIdx.x], s_sum[threadIdx.x]);
    atomicAdd(&sums[Cout + col0 + threadIdx.x], s_sq[threadIdx.x]);
  }
}

// ---------------------------------------------------------------------------
// BN finalize / masked-max aggregation
// ---------------------------------------------------------------------------
__global__ void finalize_stats_kernel(const float* __restrict__ sums,
                                      const int* __restrict__ cntPtr, int cntConst,
                                      const float* __restrict__ g, const float* __restrict__ be,
                                      float* __restrict__ scale, float* __restrict__ shift, int C) {
  int c = blockIdx.x * blockDim.x + threadIdx.x;
  if (c >= C) return;
  float cnt = cntPtr ? (float)(*cntPtr) : (float)cntConst;
  if (cnt < 1.f) cnt = 1.f;
  float mean = sums[c] / cnt;
  float var  = sums[C + c] / cnt - mean * mean;
  float sc = rsqrtf(var + EPS_) * g[c];
  scale[c] = sc;
  shift[c] = be[c] - mean * sc;
}

__global__ void bn_max_kernel(const __bf16* __restrict__ Y, const float* __restrict__ scale,
                              const float* __restrict__ shift, const int* __restrict__ mask,
                              float* __restrict__ outF, int Mtot, int C) {
  int i = blockIdx.x * blockDim.x + threadIdx.x;
  if (i >= Mtot * C) return;
  int m = i / C, c = i - (i / C) * C;
  float sc = scale[c], sh = shift[c];
  float best = -3.4e38f;
  for (int k = 0; k < K_; ++k) {
    if (mask[(size_t)m * K_ + k]) {
      float v = (float)Y[((size_t)m * K_ + k) * C + c] * sc + sh;
      if (v > best) best = v;
    }
  }
  outF[(size_t)m * C + c] = best;   // at least one neighbor always valid
}

__global__ void bn_maxpool_kernel(const __bf16* __restrict__ Y, const float* __restrict__ scale,
                                  const float* __restrict__ shift, float* __restrict__ pooled,
                                  int M, int C) {
  int i = blockIdx.x * blockDim.x + threadIdx.x;
  if (i >= B_ * C) return;
  int b = i / C, c = i - b * C;
  float sc = scale[c], sh = shift[c];
  float best = -3.4e38f;
  for (int m = 0; m < M; ++m) {
    float v = (float)Y[((size_t)(b * M + m)) * C + c] * sc + sh;
    if (v > best) best = v;
  }
  pooled[i] = best;
}

// ---------------------------------------------------------------------------
// Head MLP (8x1024 -> 512 -> 256 -> 128 -> 1), BN over batch then ReLU.
// Single workgroup, everything in LDS (48 KB).
// ---------------------------------------------------------------------------
__device__ void head_layer(float* hb, float* yb, const float* W, const float* bias,
                           const float* g, const float* be, int Cin, int Cout) {
  const int tid = threadIdx.x;
  for (int idx = tid; idx < 8 * Cout; idx += blockDim.x) {
    int r = idx / Cout, c = idx - r * Cout;
    float s = bias[c];
    const float* hr = hb + r * Cin;
    for (int k = 0; k < Cin; ++k) s += hr[k] * W[(size_t)k * Cout + c];
    yb[idx] = s;
  }
  __syncthreads();
  for (int c = tid; c < Cout; c += blockDim.x) {
    float mean = 0.f;
    for (int r = 0; r < 8; ++r) mean += yb[r * Cout + c];
    mean *= 0.125f;
    float var = 0.f;
    for (int r = 0; r < 8; ++r) { float d = yb[r * Cout + c] - mean; var += d * d; }
    var *= 0.125f;
    float sc = rsqrtf(var + EPS_) * g[c];
    float sh = be[c] - mean * sc;
    for (int r = 0; r < 8; ++r) {
      float v = yb[r * Cout + c] * sc + sh;   // BN then ReLU (head order)
      hb[r * Cout + c] = v > 0.f ? v : 0.f;
    }
  }
  __syncthreads();
}

__global__ __launch_bounds__(256) void head_kernel(
    const float* __restrict__ pooled,
    const float* W1, const float* b1, const float* g1, const float* be1,
    const float* W2, const float* b2, const float* g2, const float* be2,
    const float* W3, const float* b3, const float* g3, const float* be3,
    const float* W4, const float* b4, float* __restrict__ out) {
  __shared__ float hb[8 * 1024];
  __shared__ float yb[8 * 512];
  int tid = threadIdx.x;
  for (int i = tid; i < 8 * 1024; i += blockDim.x) hb[i] = pooled[i];
  __syncthreads();
  head_layer(hb, yb, W1, b1, g1, be1, 1024, 512);
  head_layer(hb, yb, W2, b2, g2, be2, 512, 256);
  head_layer(hb, yb, W3, b3, g3, be3, 256, 128);
  if (tid < 8) {
    float s = b4[0];
    for (int k = 0; k < 128; ++k) s += hb[tid * 128 + k] * W4[k];
    out[tid] = s;
  }
}

// ---------------------------------------------------------------------------
// Host orchestration
// ---------------------------------------------------------------------------
extern "C" void kernel_launch(void* const* d_in, const int* in_sizes, int n_in,
                              void* d_out, int out_size, void* d_ws, size_t ws_size,
                              hipStream_t stream) {
  (void)in_sizes; (void)n_in; (void)out_size; (void)ws_size;
  const float* x   = (const float*)d_in[0];
  const float* pos = (const float*)d_in[1];
  // Params flattened in setup_inputs() insertion order:
  // sa1/sa2/sa3: 3 layers each of (lin.W, lin.b, bn.g, bn.be), then
  // lin1.W, lin1.b, bn1.g, bn1.be, lin2..., bn2..., lin3..., bn3..., lin4.W, lin4.b
  const float* P[50];
  for (int i = 0; i < 50; ++i) P[i] = (const float*)d_in[i + 2];

  static const int cinArr[9]  = {4, 64, 64, 131, 128, 128, 259, 256, 512};
  static const int kpadArr[9] = {32, 64, 64, 160, 128, 128, 288, 256, 512};
  static const int coutArr[9] = {64, 64, 128, 128, 128, 256, 256, 512, 1024};

  char* base = (char*)d_ws;
  size_t off = 0;
  auto alloc = [&](size_t bytes) -> void* {
    void* p = base + off;
    off = (off + bytes + 255) & ~(size_t)255;
    return p;
  };

  int*    idx1     = (int*)alloc((size_t)B_ * M1_ * sizeof(int));
  float*  centers1 = (float*)alloc((size_t)B_ * M1_ * 3 * sizeof(float));
  int*    nbr1     = (int*)alloc((size_t)E1_ * sizeof(int));
  int*    mask1    = (int*)alloc((size_t)E1_ * sizeof(int));
  int*    idx2     = (int*)alloc((size_t)B_ * M2_ * sizeof(int));
  float*  centers2 = (float*)alloc((size_t)B_ * M2_ * 3 * sizeof(float));
  int*    nbr2     = (int*)alloc((size_t)E2_ * sizeof(int));
  int*    mask2    = (int*)alloc((size_t)E2_ * sizeof(int));
  int*    cnts     = (int*)alloc(2 * sizeof(int));
  float*  sums     = (float*)alloc(2 * 1024 * sizeof(float));
  float*  scale    = (float*)alloc(1024 * sizeof(float));
  float*  shift    = (float*)alloc(1024 * sizeof(float));
  float*  biasBuf  = (float*)alloc(1024 * sizeof(float));
  float*  sa1_out  = (float*)alloc((size_t)B_ * M1_ * 128 * sizeof(float));
  float*  sa2_out  = (float*)alloc((size_t)B_ * M2_ * 256 * sizeof(float));
  float*  pooled   = (float*)alloc((size_t)8 * 1024 * sizeof(float));
  __bf16* Xb       = (__bf16*)alloc((size_t)E1_ * 64  * sizeof(__bf16));  // 128 MB
  __bf16* Yb       = (__bf16*)alloc((size_t)E1_ * 128 * sizeof(__bf16));  // 256 MB
  __bf16* Wp[9];
  for (int l = 0; l < 9; ++l)
    Wp[l] = (__bf16*)alloc((size_t)(kpadArr[l] / 32) * (coutArr[l] / 16) * 512 * sizeof(__bf16));

  zero_u32_kernel<<<1, 64, 0, stream>>>((unsigned*)cnts, 2);

  auto run_pack = [&](int l, const float* scalePrev) {
    int total = (kpadArr[l] / 32) * (coutArr[l] / 16) * 512;
    pack_w_kernel<<<(total + 255) / 256, 256, 0, stream>>>(
        P[l * 4], Wp[l], scalePrev, cinArr[l], coutArr[l], kpadArr[l]);
  };
  auto run_fold_bias = [&](int l) {   // biasBuf = b_l + shiftPrev . W_l
    fold_bias_kernel<<<(coutArr[l] + 255) / 256, 256, 0, stream>>>(
        P[l * 4], P[l * 4 + 1], shift, biasBuf, cinArr[l], coutArr[l]);
  };
  auto run_gemm = [&](int l, const __bf16* Xin, __bf16* Yout, int E, const int* mask,
                      const float* bias) {
    int C = coutArr[l], Kp = kpadArr[l];
    zero_u32_kernel<<<(2 * C + 255) / 256, 256, 0, stream>>>((unsigned*)sums, 2 * C);
    dim3 grid(E / 128, C / 64);
    gemm_bf16_kernel<<<grid, 256, 0, stream>>>(Xin, Wp[l], bias, Yout, mask, sums, E, C, Kp);
  };
  auto run_finalize = [&](int l, const int* cntPtr, int cntConst) {
    int C = coutArr[l];
    finalize_stats_kernel<<<(C + 255) / 256, 256, 0, stream>>>(
        sums, cntPtr, cntConst, P[l * 4 + 2], P[l * 4 + 3], scale, shift, C);
  };
  // One SA layer with the previous layer's BN folded into weights/bias.
  auto run_folded_layer = [&](int l, const __bf16* Xin, __bf16* Yout, int E, const int* mask,
                              const int* cntPtr, int cntConst) {
    run_pack(l, scale);      // scale/shift hold the previous layer's BN affine
    run_fold_bias(l);
    run_gemm(l, Xin, Yout, E, mask, biasBuf);
    run_finalize(l, cntPtr, cntConst);
  };

  // First layer of each SA stage: plain weights (input features are raw).
  run_pack(0, nullptr);
  run_pack(3, nullptr);
  run_pack(6, nullptr);

  // ================= SA1 =================
  fps_kernel<<<B_, 256, 0, stream>>>(pos, N_, M1_, idx1);
  gather_centers_kernel<<<(B_ * M1_ + 255) / 256, 256, 0, stream>>>(pos, idx1, centers1, N_, B_ * M1_, M1_);
  radius_kernel<<<(B_ * M1_) / 8, 256, 0, stream>>>(pos, centers1, 0.1f * 0.1f, N_, B_ * M1_, M1_,
                                                    nbr1, mask1, cnts + 0);
  build_x1_kernel<<<(E1_ + 255) / 256, 256, 0, stream>>>(x, pos, centers1, nbr1, Xb);

  run_gemm(0, Xb, Yb, E1_, mask1, P[0 * 4 + 1]);
  run_finalize(0, cnts + 0, 0);
  run_folded_layer(1, Yb, Xb, E1_, mask1, cnts + 0, 0);
  run_folded_layer(2, Xb, Yb, E1_, mask1, cnts + 0, 0);
  bn_max_kernel<<<(B_ * M1_ * 128 + 255) / 256, 256, 0, stream>>>(Yb, scale, shift, mask1,
                                                                  sa1_out, B_ * M1_, 128);

  // ================= SA2 =================
  fps_kernel<<<B_, 256, 0, stream>>>(centers1, M1_, M2_, idx2);
  gather_centers_kernel<<<(B_ * M2_ + 255) / 256, 256, 0, stream>>>(centers1, idx2, centers2,
                                                                    M1_, B_ * M2_, M2_);
  radius_kernel<<<(B_ * M2_) / 8, 256, 0, stream>>>(centers1, centers2, 0.2f * 0.2f, M1_, B_ * M2_, M2_,
                                                    nbr2, mask2, cnts + 1);
  build_x2_kernel<<<(E2_ * 5 + 255) / 256, 256, 0, stream>>>(sa1_out, centers1, centers2, nbr2, Xb);

  run_gemm(3, Xb, Yb, E2_, mask2, P[3 * 4 + 1]);
  run_finalize(3, cnts + 1, 0);
  run_folded_layer(4, Yb, Xb, E2_, mask2, cnts + 1, 0);
  run_folded_layer(5, Xb, Yb, E2_, mask2, cnts + 1, 0);
  bn_max_kernel<<<(B_ * M2_ * 256 + 255) / 256, 256, 0, stream>>>(Yb, scale, shift, mask2,
                                                                  sa2_out, B_ * M2_, 256);

  // ================= SA3 (global) =================
  build_x3_kernel<<<(E3_ * 9 + 255) / 256, 256, 0, stream>>>(sa2_out, centers2, Xb);
  run_gemm(6, Xb, Yb, E3_, nullptr, P[6 * 4 + 1]);
  run_finalize(6, nullptr, E3_);
  run_folded_layer(7, Yb, Xb, E3_, nullptr, nullptr, E3_);
  run_folded_layer(8, Xb, Yb, E3_, nullptr, nullptr, E3_);
  bn_maxpool_kernel<<<(B_ * 1024 + 255) / 256, 256, 0, stream>>>(Yb, scale, shift, pooled, M2_, 1024);

  // ================= Head =================
  head_kernel<<<1, 256, 0, stream>>>(pooled,
                                     P[36], P[37], P[38], P[39],
                                     P[40], P[41], P[42], P[43],
                                     P[44], P[45], P[46], P[47],
                                     P[48], P[49], (float*)d_out);
}
